// SequenceProfileAttention_63118839382523
// MI455X (gfx1250) — matrix operand order repaired and checked by reading, not compiled, not run on hardware
//
#include <hip/hip_runtime.h>
#include <hip/hip_bf16.h>

// ---------------------------------------------------------------------------
// Problem constants (match reference)
// ---------------------------------------------------------------------------
#define BB   64
#define LL   512
#define DD   512
#define HH   4
#define HD   128
#define NAA  20
#define FF   2048

typedef _Float16 h8v  __attribute__((ext_vector_type(8)));
typedef _Float16 h16v __attribute__((ext_vector_type(16)));
typedef float    f8v  __attribute__((ext_vector_type(8)));

// epilogue flags
#define GF_RELU     1
#define GF_OUT_F32  2
#define GF_OUT_F16  4
#define GF_OUT_F16T 8
#define GF_RES      16
#define GF_BIAS     32

// ---------------------------------------------------------------------------
// Generic batched WMMA GEMM:  C[M,N] = alpha * A[M,K] @ Bt[N,K]^T (+bias)(+res)
// A, Bt are f16.  One 32x32 register tile (2x2 WMMA tiles) per wave,
// 4 waves per block covering 128 columns.  Batch index z decomposes as (b, h)
// with per-b and per-h strides for A/B/C.
// Fragment layouts per CDNA5 ISA 7.12.2 (16-bit A 16x32 / B 32x16 / f32 C 16x16)
// ---------------------------------------------------------------------------
__global__ __launch_bounds__(128)
void wmma_gemm_kernel(const _Float16* __restrict__ A,  long lda, long sAb, long sAh,
                      const _Float16* __restrict__ Bt, long ldb, long sBb, long sBh,
                      float* __restrict__ Cf, _Float16* __restrict__ Ch,
                      long ldc, long sCb, long sCh,
                      const float* __restrict__ bias,
                      const float* __restrict__ res, long ldres,
                      float alpha, int M, int N, int K, int H, int flags)
{
    const int wave = threadIdx.x >> 5;
    const int lane = threadIdx.x & 31;
    const int nt   = blockIdx.x * 4 + wave;   // 32-column tile index
    const int mt   = blockIdx.y;              // 32-row tile index
    if (nt * 32 >= N) return;                 // uniform per wave -> EXEC stays all-ones

    const int z = blockIdx.z;
    const int b = z / H;
    const int h = z - b * H;

    const int half = lane >> 4;               // which 16-lane half
    const int r    = lane & 15;               // row (A) / column (B) / N-index (C)

    const _Float16* Ab = A  + (size_t)b * sAb + (size_t)h * sAh;
    const _Float16* Bb = Bt + (size_t)b * sBb + (size_t)h * sBh;

    const _Float16* arow[2];
    const _Float16* bcol[2];
    arow[0] = Ab + (size_t)(mt * 32 + r) * lda;
    arow[1] = Ab + (size_t)(mt * 32 + 16 + r) * lda;
    bcol[0] = Bb + (size_t)(nt * 32 + r) * ldb;
    bcol[1] = Bb + (size_t)(nt * 32 + 16 + r) * ldb;

    f8v acc[2][2];
    #pragma unroll
    for (int mi = 0; mi < 2; ++mi)
        #pragma unroll
        for (int ni = 0; ni < 2; ++ni)
            #pragma unroll
            for (int i = 0; i < 8; ++i) acc[mi][ni][i] = 0.0f;

    for (int kk = 0; kk < K; kk += 32) {
        // WGP-scope prefetch two K-steps ahead (global_prefetch_b8 scope:WGP
        // pulls into all cache levels); speculative, OOB-safe.
        if (kk + 64 < K) {
            __builtin_prefetch(arow[0] + kk + 64, 0, 3);
            __builtin_prefetch(bcol[0] + kk + 64, 0, 3);
        }
        h16v af[2], bf[2];
        // A 16x32 f16: lanes 0-15 hold K{0..7,16..23}, lanes 16-31 K{8..15,24..31}
        #pragma unroll
        for (int mi = 0; mi < 2; ++mi) {
            const _Float16* ar = arow[mi] + kk;
            *((h8v*)&af[mi])     = *(const h8v*)(ar + half * 8);
            *((h8v*)&af[mi] + 1) = *(const h8v*)(ar + 16 + half * 8);
        }
        // B 32x16 f16: lanes 0-15 hold K{0..7,16..23}, lanes 16-31 K{8..15,24..31} of column r
        #pragma unroll
        for (int ni = 0; ni < 2; ++ni) {
            const _Float16* br = bcol[ni] + kk;
            *((h8v*)&bf[ni])     = *(const h8v*)(br + half * 8);
            *((h8v*)&bf[ni] + 1) = *(const h8v*)(br + 16 + half * 8);
        }
        acc[0][0] = __builtin_amdgcn_wmma_f32_16x16x32_f16(false, af[0], false, bf[0], (short)0, acc[0][0], false, false);
        acc[0][1] = __builtin_amdgcn_wmma_f32_16x16x32_f16(false, af[0], false, bf[1], (short)0, acc[0][1], false, false);
        acc[1][0] = __builtin_amdgcn_wmma_f32_16x16x32_f16(false, af[1], false, bf[0], (short)0, acc[1][0], false, false);
        acc[1][1] = __builtin_amdgcn_wmma_f32_16x16x32_f16(false, af[1], false, bf[1], (short)0, acc[1][1], false, false);
    }

    #pragma unroll
    for (int mi = 0; mi < 2; ++mi) {
        #pragma unroll
        for (int ni = 0; ni < 2; ++ni) {
            const int col = nt * 32 + ni * 16 + r;
            const float bv = (flags & GF_BIAS) ? bias[col] : 0.0f;
            const size_t cb = (size_t)b * sCb + (size_t)h * sCh;
            #pragma unroll
            for (int i = 0; i < 8; ++i) {
                const int row = mt * 32 + mi * 16 + i + half * 8;  // C: VGPR i -> M=i (+8 hi half)
                float v = acc[mi][ni][i] * alpha + bv;
                if (flags & GF_RES)  v += res[(size_t)row * ldres + col];
                if (flags & GF_RELU) v = v > 0.0f ? v : 0.0f;
                if (flags & GF_OUT_F32)  Cf[cb + (size_t)row * ldc + col] = v;
                if (flags & GF_OUT_F16)  Ch[cb + (size_t)row * ldc + col] = (_Float16)v;
                if (flags & GF_OUT_F16T) Ch[cb + (size_t)col * ldc + row] = (_Float16)v;
            }
        }
    }
}

// ---------------------------------------------------------------------------
// Weight transpose + f32->f16 convert:  W[K,N] (row major) -> Wt[N,K] f16
// ---------------------------------------------------------------------------
__global__ void convert_wt_kernel(const float* __restrict__ W, _Float16* __restrict__ Wt,
                                  int K, int N)
{
    size_t idx = (size_t)blockIdx.x * blockDim.x + threadIdx.x;
    if (idx >= (size_t)K * N) return;
    int k = (int)(idx / N);
    int n = (int)(idx - (size_t)k * N);
    Wt[(size_t)n * K + k] = (_Float16)W[idx];
}

// ---------------------------------------------------------------------------
// LayerNorm over D=512 (one block of 256 per row), optional affine, f16 out
// ---------------------------------------------------------------------------
__global__ __launch_bounds__(256)
void layernorm_kernel(const float* __restrict__ in, _Float16* __restrict__ outh,
                      const float* __restrict__ g, const float* __restrict__ beta)
{
    const size_t row = blockIdx.x;
    const float* x = in + row * DD;
    const int t = threadIdx.x;
    float v0 = x[t], v1 = x[t + 256];

    __shared__ float rs1[256], rs2[256];
    rs1[t] = v0 + v1;
    rs2[t] = v0 * v0 + v1 * v1;
    __syncthreads();
    for (int s = 128; s > 0; s >>= 1) {
        if (t < s) { rs1[t] += rs1[t + s]; rs2[t] += rs2[t + s]; }
        __syncthreads();
    }
    const float mean = rs1[0] * (1.0f / DD);
    const float var  = rs2[0] * (1.0f / DD) - mean * mean;
    const float rstd = rsqrtf(var + 1e-5f);

    float y0 = (v0 - mean) * rstd;
    float y1 = (v1 - mean) * rstd;
    if (g) { y0 = y0 * g[t] + beta[t]; y1 = y1 * g[t + 256] + beta[t + 256]; }
    outh[row * DD + t]       = (_Float16)y0;
    outh[row * DD + t + 256] = (_Float16)y1;
}

// ---------------------------------------------------------------------------
// Profile head: logits = sln @ wp + bp ; softmax ; entropy (per row)
// ---------------------------------------------------------------------------
__global__ __launch_bounds__(128)
void profile_kernel(const _Float16* __restrict__ sln, const float* __restrict__ wp,
                    const float* __restrict__ bp, float* __restrict__ probs_out,
                    float* __restrict__ ent_out)
{
    const size_t row = blockIdx.x;
    __shared__ float xs[DD];
    __shared__ float lg[32];
    const int t = threadIdx.x;

    for (int i = t; i < DD; i += 128) xs[i] = (float)sln[row * DD + i];
    __syncthreads();

    if (t < NAA) {
        float acc = bp[t];
        for (int d = 0; d < DD; ++d) acc += xs[d] * wp[d * NAA + t];
        lg[t] = acc;
    }
    __syncthreads();

    if (t == 0) {
        float m = -1e30f;
        for (int j = 0; j < NAA; ++j) m = fmaxf(m, lg[j]);
        float s = 0.0f;
        for (int j = 0; j < NAA; ++j) { lg[j] = __expf(lg[j] - m); s += lg[j]; }
        const float inv = 1.0f / s;
        float e = 0.0f;
        for (int j = 0; j < NAA; ++j) {
            const float p = lg[j] * inv;
            lg[j] = p;
            e -= p * __logf(p + 1e-8f);
        }
        ent_out[row] = e;
    }
    __syncthreads();
    if (t < NAA) probs_out[row * NAA + t] = lg[t];
}

// ---------------------------------------------------------------------------
// total_bias[l] = -entropy[0,l] * ew + pos_bias[l]
// ---------------------------------------------------------------------------
__global__ void bias_kernel(const float* __restrict__ ent, const float* __restrict__ ew,
                            const float* __restrict__ pos, float* __restrict__ tb)
{
    const int i = threadIdx.x + blockIdx.x * blockDim.x;
    if (i < LL) tb[i] = -ent[i] * ew[0] + pos[i];
}

// ---------------------------------------------------------------------------
// Attention softmax over 512 keys (f16 in-place); adds bias at query l==0;
// head 0 probabilities exported to d_out in f32.
// ---------------------------------------------------------------------------
__global__ __launch_bounds__(128)
void attn_softmax_kernel(_Float16* __restrict__ scores, const float* __restrict__ tb,
                         float* __restrict__ attn0_out)
{
    const int l = blockIdx.x, h = blockIdx.y, b = blockIdx.z;
    _Float16* rowp = scores + (((size_t)(b * HH + h)) * LL + l) * LL;
    const int t = threadIdx.x;

    __shared__ float red[128];
    float x[4];
    float mx = -1e30f;
    #pragma unroll
    for (int i = 0; i < 4; ++i) {
        const int m = t + i * 128;
        float v = (float)rowp[m];
        if (l == 0) v += tb[m];
        x[i] = v;
        mx = fmaxf(mx, v);
    }
    red[t] = mx; __syncthreads();
    for (int s = 64; s > 0; s >>= 1) { if (t < s) red[t] = fmaxf(red[t], red[t + s]); __syncthreads(); }
    mx = red[0]; __syncthreads();

    float sum = 0.0f;
    #pragma unroll
    for (int i = 0; i < 4; ++i) { x[i] = __expf(x[i] - mx); sum += x[i]; }
    red[t] = sum; __syncthreads();
    for (int s = 64; s > 0; s >>= 1) { if (t < s) red[t] += red[t + s]; __syncthreads(); }
    const float inv = 1.0f / red[0];

    #pragma unroll
    for (int i = 0; i < 4; ++i) {
        const int m = t + i * 128;
        const float p = x[i] * inv;
        rowp[m] = (_Float16)p;
        if (h == 0) attn0_out[((size_t)b * LL + l) * LL + m] = p;
    }
}

// ---------------------------------------------------------------------------
// Host-side launch helper
// ---------------------------------------------------------------------------
static void launch_gemm(hipStream_t stream,
                        const _Float16* A, long lda, long sAb, long sAh,
                        const _Float16* Bt, long ldb, long sBb, long sBh,
                        float* Cf, _Float16* Ch, long ldc, long sCb, long sCh,
                        const float* bias, const float* res, long ldres,
                        float alpha, int M, int N, int K, int H, int batch, int flags)
{
    dim3 grid((N + 127) / 128, M / 32, batch);
    wmma_gemm_kernel<<<grid, 128, 0, stream>>>(A, lda, sAb, sAh, Bt, ldb, sBb, sBh,
                                               Cf, Ch, ldc, sCb, sCh, bias, res, ldres,
                                               alpha, M, N, K, H, flags);
}

static inline size_t align256(size_t x) { return (x + 255) & ~(size_t)255; }

// ---------------------------------------------------------------------------
// kernel_launch
// ---------------------------------------------------------------------------
extern "C" void kernel_launch(void* const* d_in, const int* in_sizes, int n_in,
                              void* d_out, int out_size, void* d_ws, size_t ws_size,
                              hipStream_t stream)
{
    (void)in_sizes; (void)n_in; (void)out_size; (void)ws_size;

    const float* s   = (const float*)d_in[0];
    const float* z   = (const float*)d_in[1];
    const float* wq  = (const float*)d_in[2];
    const float* bq  = (const float*)d_in[3];
    const float* wk  = (const float*)d_in[4];
    const float* bk  = (const float*)d_in[5];
    const float* wv  = (const float*)d_in[6];
    const float* bv  = (const float*)d_in[7];
    const float* wo  = (const float*)d_in[8];
    const float* bo  = (const float*)d_in[9];
    const float* wp  = (const float*)d_in[10];
    const float* bp  = (const float*)d_in[11];
    const float* pos = (const float*)d_in[12];
    const float* ew  = (const float*)d_in[13];
    const float* lng = (const float*)d_in[14];
    const float* lnb = (const float*)d_in[15];
    const float* w1  = (const float*)d_in[16];
    const float* b1  = (const float*)d_in[17];
    const float* w2  = (const float*)d_in[18];
    const float* b2  = (const float*)d_in[19];

    float* out = (float*)d_out;
    const size_t n_s3   = (size_t)BB * LL * DD;     // 16,777,216
    const size_t n_z    = (size_t)BB * LL * 64;     //  2,097,152
    const size_t n_attn = (size_t)BB * LL * LL;     // 16,777,216
    const size_t n_ent  = (size_t)BB * LL;          //     32,768
    float* out_s3   = out;
    float* out_z    = out + n_s3;
    float* out_attn = out + n_s3 + n_z;
    float* out_ent  = out + n_s3 + n_z + n_attn;
    float* out_prob = out + n_s3 + n_z + n_attn + n_ent;

    // ---- workspace layout --------------------------------------------------
    char* ws = (char*)d_ws;
    size_t off = 0;
    const size_t ROWS = (size_t)BB * LL;            // 32768

    _Float16* wqt = (_Float16*)(ws + off); off = align256(off + (size_t)DD * DD * 2);
    _Float16* wkt = (_Float16*)(ws + off); off = align256(off + (size_t)DD * DD * 2);
    _Float16* wvt = (_Float16*)(ws + off); off = align256(off + (size_t)DD * DD * 2);
    _Float16* wot = (_Float16*)(ws + off); off = align256(off + (size_t)DD * DD * 2);
    _Float16* w1t = (_Float16*)(ws + off); off = align256(off + (size_t)DD * FF * 2);
    _Float16* w2t = (_Float16*)(ws + off); off = align256(off + (size_t)FF * DD * 2);
    _Float16* sln = (_Float16*)(ws + off); off = align256(off + ROWS * DD * 2);       // reused as h
    _Float16* qh  = (_Float16*)(ws + off); off = align256(off + ROWS * DD * 2);       // reused as attn_out
    _Float16* kh  = (_Float16*)(ws + off); off = align256(off + ROWS * DD * 2);
    _Float16* vT  = (_Float16*)(ws + off); off = align256(off + ROWS * DD * 2);       // [b,h,d,m]
    _Float16* sc  = (_Float16*)(ws + off); off = align256(off + (size_t)BB * HH * LL * LL * 2); // reused as ffn1
    float*    s2  = (float*)   (ws + off); off = align256(off + ROWS * DD * 4);
    float*    tb  = (float*)   (ws + off); off = align256(off + LL * 4);

    _Float16* hh    = sln;   // reuse
    _Float16* aoh   = qh;    // reuse
    _Float16* ffn1h = sc;    // reuse (B*L*2048 == B*H*L*L elements)

    // ---- 1) weight convert/transpose  W[K,N] -> Wt[N,K] f16 ---------------
    {
        const int TPB = 256;
        size_t n;
        n = (size_t)DD * DD;
        convert_wt_kernel<<<(n + TPB - 1) / TPB, TPB, 0, stream>>>(wq, wqt, DD, DD);
        convert_wt_kernel<<<(n + TPB - 1) / TPB, TPB, 0, stream>>>(wk, wkt, DD, DD);
        convert_wt_kernel<<<(n + TPB - 1) / TPB, TPB, 0, stream>>>(wv, wvt, DD, DD);
        convert_wt_kernel<<<(n + TPB - 1) / TPB, TPB, 0, stream>>>(wo, wot, DD, DD);
        n = (size_t)DD * FF;
        convert_wt_kernel<<<(n + TPB - 1) / TPB, TPB, 0, stream>>>(w1, w1t, DD, FF);
        convert_wt_kernel<<<(n + TPB - 1) / TPB, TPB, 0, stream>>>(w2, w2t, FF, DD);
    }

    // ---- 2) layernorm(s) -> sln (f16) -------------------------------------
    layernorm_kernel<<<ROWS, 256, 0, stream>>>(s, sln, nullptr, nullptr);

    // ---- 3) profile probs + entropy ---------------------------------------
    profile_kernel<<<ROWS, 128, 0, stream>>>(sln, wp, bp, out_prob, out_ent);

    // ---- 4) total bias from batch-0 entropy -------------------------------
    bias_kernel<<<(LL + 255) / 256, 256, 0, stream>>>(out_ent, ew, pos, tb);

    // ---- 5) Q, K projections (f16 out) ------------------------------------
    launch_gemm(stream, sln, DD, 0, 0, wqt, DD, 0, 0, nullptr, qh, DD, 0, 0,
                bq, nullptr, 0, 1.0f, (int)ROWS, DD, DD, 1, 1, GF_OUT_F16 | GF_BIAS);
    launch_gemm(stream, sln, DD, 0, 0, wkt, DD, 0, 0, nullptr, kh, DD, 0, 0,
                bk, nullptr, 0, 1.0f, (int)ROWS, DD, DD, 1, 1, GF_OUT_F16 | GF_BIAS);

    // ---- 6) V projection, written transposed per head: vT[b, h*hd+d, m] ---
    launch_gemm(stream, sln, DD, (long)LL * DD, 0, wvt, DD, 0, 0,
                nullptr, vT, LL, (long)DD * LL, 0,
                bv, nullptr, 0, 1.0f, LL, DD, DD, 1, BB, GF_OUT_F16T | GF_BIAS);

    // ---- 7) scores[b,h,l,m] = q . k / sqrt(hd)  (f16) ----------------------
    launch_gemm(stream, qh, DD, (long)LL * DD, HD, kh, DD, (long)LL * DD, HD,
                nullptr, sc, LL, (long)HH * LL * LL, (long)LL * LL,
                nullptr, nullptr, 0, 0.088388347648318447f /* 1/sqrt(128) */,
                LL, LL, HD, HH, BB * HH, GF_OUT_F16);

    // ---- 8) softmax (+ row-0 bias) in-place; export head-0 probs ----------
    {
        dim3 grid(LL, HH, BB);
        attn_softmax_kernel<<<grid, 128, 0, stream>>>(sc, tb, out_attn);
    }

    // ---- 9) attn_out[b,l,h*hd+d] = probs @ vT ------------------------------
    launch_gemm(stream, sc, LL, (long)HH * LL * LL, (long)LL * LL,
                vT, LL, (long)DD * LL, (long)HD * LL,
                nullptr, aoh, DD, (long)LL * DD, HD,
                nullptr, nullptr, 0, 1.0f, LL, HD, LL, HH, BB * HH, GF_OUT_F16);

    // ---- 10) s2 = s + attn_out @ wo + bo  (f32) ----------------------------
    launch_gemm(stream, aoh, DD, 0, 0, wot, DD, 0, 0, s2, nullptr, DD, 0, 0,
                bo, s, DD, 1.0f, (int)ROWS, DD, DD, 1, 1,
                GF_OUT_F32 | GF_BIAS | GF_RES);

    // ---- 11) h = LN(s2)*g + b  (f16) ---------------------------------------
    layernorm_kernel<<<ROWS, 256, 0, stream>>>(s2, hh, lng, lnb);

    // ---- 12) ffn1 = relu(h @ w1 + b1) (f16) --------------------------------
    launch_gemm(stream, hh, DD, 0, 0, w1t, DD, 0, 0, nullptr, ffn1h, FF, 0, 0,
                b1, nullptr, 0, 1.0f, (int)ROWS, FF, DD, 1, 1,
                GF_OUT_F16 | GF_BIAS | GF_RELU);

    // ---- 13) s3 = s2 + ffn1 @ w2 + b2 -> d_out ------------------------------
    launch_gemm(stream, ffn1h, FF, 0, 0, w2t, FF, 0, 0, out_s3, nullptr, DD, 0, 0,
                b2, s2, DD, 1.0f, (int)ROWS, DD, FF, 1, 1,
                GF_OUT_F32 | GF_BIAS | GF_RES);

    // ---- 14) z passthrough --------------------------------------------------
    hipMemcpyAsync(out_z, z, n_z * sizeof(float), hipMemcpyDeviceToDevice, stream);
}
